// Token_performer_71932112273751
// MI455X (gfx1250) — compile-verified, matrix operand order checked by reading, not run on hardware
//
#include <hip/hip_runtime.h>
#include <cmath>

typedef __bf16 bf16_t;
typedef __attribute__((ext_vector_type(16))) __bf16 v16bf;
typedef __attribute__((ext_vector_type(8)))  __bf16 v8bf;
typedef __attribute__((ext_vector_type(8)))  float  v8f;
typedef unsigned int u32;
typedef __attribute__((ext_vector_type(4))) u32 v4u;
typedef __attribute__((ext_vector_type(4))) int v4i;
typedef __attribute__((ext_vector_type(8))) int v8i;

#define B_    8
#define T_    4096
#define ROWS_ (B_ * T_)      // 32768
#define DIM_  512
#define EMB_  512
#define M_    256

// ---------------- wave32 reduction ----------------
__device__ __forceinline__ float wred(float v) {
#pragma unroll
    for (int o = 16; o > 0; o >>= 1) v += __shfl_xor(v, o, 32);
    return v;
}

// ---------------- fp32 -> bf16 convert ----------------
__global__ __launch_bounds__(256)
void f2bf_kernel(const float* __restrict__ in, bf16_t* __restrict__ out, int n) {
    int i = blockIdx.x * 256 + threadIdx.x;
    if (i < n) out[i] = (bf16_t)in[i];
}

// ---------------- LayerNorm (one wave per row of C floats) ----------------
__global__ __launch_bounds__(256)
void ln_kernel(const float* __restrict__ x, const float* __restrict__ g,
               const float* __restrict__ be, bf16_t* __restrict__ out, int C) {
    int lane = threadIdx.x & 31, wave = threadIdx.x >> 5;
    long row = (long)blockIdx.x * 8 + wave;
    const float* xr = x + row * C;
    float s = 0.f, ss = 0.f;
    for (int i = lane; i < C; i += 32) { float v = xr[i]; s += v; ss += v * v; }
    s = wred(s); ss = wred(ss);
    float mu = s / C;
    float rstd = rsqrtf(ss / C - mu * mu + 1e-5f);
    bf16_t* orow = out + row * C;
    for (int i = lane; i < C; i += 32)
        orow[i] = (bf16_t)((xr[i] - mu) * rstd * g[i] + be[i]);
}

// ---------------- 0.5*||k||^2, 0.5*||q||^2 from bf16 kqv rows ----------------
__global__ __launch_bounds__(256)
void sumsq_kernel(const bf16_t* __restrict__ kqv, float* __restrict__ xdk,
                  float* __restrict__ xdq) {
    int lane = threadIdx.x & 31, wave = threadIdx.x >> 5;
    long row = (long)blockIdx.x * 8 + wave;
    const bf16_t* r = kqv + row * (3 * EMB_);
    float sk = 0.f, sq = 0.f;
    for (int i = lane; i < EMB_; i += 32) {
        float a = (float)r[i];        sk += a * a;
        float b = (float)r[EMB_ + i]; sq += b * b;
    }
    sk = wred(sk); sq = wred(sq);
    if (lane == 0) { xdk[row] = 0.5f * sk; xdq[row] = 0.5f * sq; }
}

// ---------------- ksum[b,m] = sum_t kp[b,t,m] ----------------
__global__ __launch_bounds__(256)
void ksum_kernel(const bf16_t* __restrict__ kp, float* __restrict__ ksum) {
    int idx = blockIdx.x * 256 + threadIdx.x;   // b*256 + m
    int b = idx >> 8, m = idx & 255;
    const bf16_t* p = kp + (long)b * T_ * M_ + m;
    float s = 0.f;
    for (int t = 0; t < T_; t++) s += (float)p[(long)t * M_];
    ksum[idx] = s;
}

// ---------------- Dinv[t] = 1/(qp[t,:]·ksum[b,:] + eps) ----------------
__global__ __launch_bounds__(256)
void dinv_kernel(const bf16_t* __restrict__ qp, const float* __restrict__ ksum,
                 float* __restrict__ dinv) {
    int lane = threadIdx.x & 31, wave = threadIdx.x >> 5;
    long row = (long)blockIdx.x * 8 + wave;
    int b = (int)(row >> 12);
    const bf16_t* q = qp + row * M_;
    const float* ks = ksum + b * M_;
    float s = 0.f;
    for (int i = lane; i < M_; i += 32) s += (float)q[i] * ks[i];
    s = wred(s);
    if (lane == 0) dinv[row] = 1.0f / (s + 1e-8f);
}

// ---------------- bf16 tile transpose: out[c, r] = in[r, c] (per batch z) ----
__global__ __launch_bounds__(256)
void transpose_bf16_kernel(const bf16_t* __restrict__ in, long inBatch, int ldin,
                           bf16_t* __restrict__ out, long outBatch, int R) {
    __shared__ bf16_t tile[32][33];
    const bf16_t* I = in + (long)blockIdx.z * inBatch;
    bf16_t* O = out + (long)blockIdx.z * outBatch;
    int c0 = blockIdx.x * 32, r0 = blockIdx.y * 32;
    int tx = threadIdx.x & 31, ty = threadIdx.x >> 5;   // 32 x 8
#pragma unroll
    for (int i = 0; i < 4; i++) {
        int r = r0 + ty + i * 8;
        tile[ty + i * 8][tx] = I[(long)r * ldin + c0 + tx];
    }
    __syncthreads();
#pragma unroll
    for (int i = 0; i < 4; i++) {
        int c = c0 + ty + i * 8;
        O[(long)c * R + r0 + tx] = tile[tx][ty + i * 8];
    }
}

// ---------------- WMMA helper ----------------
__device__ __forceinline__ v8f wmma_bf16(v16bf a, v16bf b, v8f c) {
    return __builtin_amdgcn_wmma_f32_16x16x32_bf16(false, a, false, b,
                                                   (short)0, c, false, false);
}

enum { MODE_KQV = 0, MODE_PRM = 1, MODE_ATT = 2, MODE_PROJ = 3,
       MODE_MLP1 = 4, MODE_MLP2 = 5, MODE_PLAIN = 6 };

#define KSTEP 64
#define LSTR  72   // 64 data elems + 8 pad elems per row (144B, 16B-aligned)
#define ATILE (128 * LSTR)
#define BTILE (128 * LSTR)

// TDM: DMA one 2D bf16 tile (rows x KSTEP, row stride = ldElems) from global
// into LDS at ldsAddr, padding each 128B row with 16B (matches LSTR layout).
// Issued once per wave that calls it; tracked by TENSORcnt.
__device__ __forceinline__ void tdm_load_tile(const bf16_t* gaddr, u32 ldsAddr,
                                              int rows, int ldElems) {
    unsigned long long ga = (unsigned long long)(size_t)gaddr;
    v4u g0;
    g0[0] = 1u;                                     // count=1 (valid user D#)
    g0[1] = ldsAddr;                                // lds_addr (bytes)
    g0[2] = (u32)ga;                                // global_addr[31:0]
    g0[3] = (u32)((ga >> 32) & 0x01ffffffu) | (2u << 30);  // addr[56:32], type=2
    // group1: data_size=2B, pad_enable, pad_interval=32 DW (128B), pad_amount=4 DW
    u32 d0  = (1u << 16) | (1u << 20) | (4u << 22) | (3u << 25);
    u32 td0 = 0x40000000u;                          // huge tensor dims: no OOB clip
    u32 td1 = 0x40000000u;
    u32 tl0 = (u32)KSTEP;                           // tile_dim0 (elems per row)
    u32 tl1 = (u32)rows;                            // tile_dim1 (rows)
    unsigned long long s0 = (unsigned long long)(u32)ldElems;  // dim0 stride (elems)
    v8i g1;
    g1[0] = (int)d0;
    g1[1] = (int)((td0 & 0xffffu) << 16);                       // [63:48]=td0 lo
    g1[2] = (int)(((td0 >> 16) & 0xffffu) | ((td1 & 0xffffu) << 16));
    g1[3] = (int)(((td1 >> 16) & 0xffffu) | (tl0 << 16));       // [127:112]=tile0
    g1[4] = (int)tl1;                                           // tile1, tile2=0
    g1[5] = (int)(u32)s0;                                       // stride0 lo
    g1[6] = (int)(u32)((s0 >> 32) & 0xffffu);                   // stride0 hi, s1 lo=0
    g1[7] = 0;
    v4i z4 = {0, 0, 0, 0};
    v8i z8 = {0, 0, 0, 0, 0, 0, 0, 0};
    __builtin_amdgcn_tensor_load_to_lds(g0, g1, z4, z4, z8, 0);
}

// C[row,col] = sum_k A[row,k] * W[col,k]  (A:[M,K] bf16, W:[N,K] bf16)
// 256 thr = 8 waves (2 row-groups x 4 col-groups); block tile 128x128;
// wave tile 64x32 (c[4][2]); K staged 64 at a time via TDM, double-buffered.
template <int MODE>
__global__ __launch_bounds__(256)
void gemm_kernel(const bf16_t* __restrict__ A, int lda,
                 const bf16_t* __restrict__ W, int ldw, long wBatchStride,
                 const float* __restrict__ bias, int K,
                 bf16_t* __restrict__ outB, int ldo,
                 float* __restrict__ outF,
                 const float* __restrict__ rowAux,
                 const float* __restrict__ addSrc) {
    __shared__ alignas(16) bf16_t sm[2 * ATILE + 2 * BTILE];

    const int tid  = threadIdx.x;
    const int lane = tid & 31;
    const int wave = tid >> 5;
    const int half = lane >> 4;
    const int l15  = lane & 15;
    const int rm   = (wave & 1) * 64;            // wave row base in tile
    const int cn   = (wave >> 1) * 32;           // wave col base in tile
    const int row0 = blockIdx.y * 128;
    const int col0 = blockIdx.x * 128;
    const bf16_t* Ab = A + (long)row0 * lda;
    const bf16_t* Wb = W + (long)col0 * ldw +
                       (wBatchStride ? (long)(row0 >> 12) * wBatchStride : 0);

    // wave 0 DMA-stages one K-tile (A:128x64, B:128x64) into LDS buffer s
    auto stage = [&](int s, int k0) {
        tdm_load_tile(Ab + k0, (u32)(size_t)(sm + s * ATILE), 128, lda);
        tdm_load_tile(Wb + k0, (u32)(size_t)(sm + 2 * ATILE + s * BTILE), 128, ldw);
    };

    v8f c[4][2] = {};
    int cur = 0;
    if (wave == 0) stage(cur, 0);
    for (int k0 = 0; k0 < K; k0 += KSTEP) {
        if (wave == 0) __builtin_amdgcn_s_wait_tensorcnt((short)0);
        __syncthreads();                 // cur tile staged; prev reads finished
        if (wave == 0 && k0 + KSTEP < K) stage(cur ^ 1, k0 + KSTEP);

        const bf16_t* aCur = sm + cur * ATILE;
        const bf16_t* bCur = sm + 2 * ATILE + cur * BTILE;
#pragma unroll
        for (int ks = 0; ks < KSTEP; ks += 32) {
            v16bf a[4], b[2];
#pragma unroll
            for (int i = 0; i < 4; i++) {
                const bf16_t* ap = aCur + (rm + i * 16 + l15) * LSTR + ks + half * 8;
                ((v8bf*)&a[i])[0] = *(const v8bf*)ap;
                ((v8bf*)&a[i])[1] = *(const v8bf*)(ap + 16);
            }
#pragma unroll
            for (int j = 0; j < 2; j++) {
                const bf16_t* bp = bCur + (cn + j * 16 + l15) * LSTR + ks + half * 8;
                ((v8bf*)&b[j])[0] = *(const v8bf*)bp;
                ((v8bf*)&b[j])[1] = *(const v8bf*)(bp + 16);
            }
#pragma unroll
            for (int i = 0; i < 4; i++)
#pragma unroll
            for (int j = 0; j < 2; j++)
                c[i][j] = wmma_bf16(a[i], b[j], c[i][j]);
        }
        cur ^= 1;
    }

#pragma unroll
    for (int i = 0; i < 4; i++)
#pragma unroll
    for (int j = 0; j < 2; j++) {
        int col = col0 + cn + j * 16 + l15;
#pragma unroll
        for (int r = 0; r < 8; r++) {
            long row = row0 + rm + i * 16 + r + half * 8;
            float acc = c[i][j][r];
            if constexpr (MODE == MODE_KQV) {
                float g = acc + bias[col];
                outB[row * ldo + col] = (bf16_t)g;
                if (col >= 2 * EMB_)
                    outF[row * EMB_ + (col - 2 * EMB_)] = g;   // fp32 v for residual
            } else if constexpr (MODE == MODE_PRM) {
                outB[row * ldo + col] =
                    (bf16_t)(expf(acc - rowAux[row]) * 0.0625f); // 1/sqrt(256)
            } else if constexpr (MODE == MODE_ATT) {
                outB[row * ldo + col] = (bf16_t)(acc * rowAux[row]);
            } else if constexpr (MODE == MODE_PROJ) {
                outF[row * ldo + col] = addSrc[row * EMB_ + col] + acc + bias[col];
            } else if constexpr (MODE == MODE_MLP1) {
                float t = acc + bias[col];
                outB[row * ldo + col] =
                    (bf16_t)(0.5f * t * (1.0f + erff(t * 0.70710678118f)));
            } else if constexpr (MODE == MODE_MLP2) {
                outF[row * ldo + col] = addSrc[row * EMB_ + col] + acc + bias[col];
            } else {  // MODE_PLAIN
                outB[row * ldo + col] = (bf16_t)acc;
            }
        }
    }
}

extern "C" void kernel_launch(void* const* d_in, const int* in_sizes, int n_in,
                              void* d_out, int out_size, void* d_ws, size_t ws_size,
                              hipStream_t stream) {
    (void)in_sizes; (void)n_in; (void)out_size; (void)ws_size;
    const float* x      = (const float*)d_in[0];
    const float* w_kqv  = (const float*)d_in[1];
    const float* b_kqv  = (const float*)d_in[2];
    const float* w_proj = (const float*)d_in[3];
    const float* b_proj = (const float*)d_in[4];
    const float* g1     = (const float*)d_in[5];
    const float* be1    = (const float*)d_in[6];
    const float* g2     = (const float*)d_in[7];
    const float* be2    = (const float*)d_in[8];
    const float* w_mlp1 = (const float*)d_in[9];
    const float* b_mlp1 = (const float*)d_in[10];
    const float* w_mlp2 = (const float*)d_in[11];
    const float* b_mlp2 = (const float*)d_in[12];
    const float* w_perf = (const float*)d_in[13];
    float* out = (float*)d_out;

    char* ws = (char*)d_ws;
    size_t off = 0;
    auto carve = [&](size_t bytes) -> char* {
        char* p = ws + off;
        off = (off + bytes + 255) & ~(size_t)255;
        return p;
    };

    bf16_t* wkqv_b  = (bf16_t*)carve((size_t)3 * EMB_ * DIM_ * 2);
    bf16_t* wperf_b = (bf16_t*)carve((size_t)M_ * EMB_ * 2);
    bf16_t* wproj_b = (bf16_t*)carve((size_t)EMB_ * EMB_ * 2);
    bf16_t* wm1_b   = (bf16_t*)carve((size_t)EMB_ * EMB_ * 2);
    bf16_t* wm2_b   = (bf16_t*)carve((size_t)EMB_ * EMB_ * 2);
    bf16_t* xn_b    = (bf16_t*)carve((size_t)ROWS_ * DIM_ * 2);
    bf16_t* kqv_b   = (bf16_t*)carve((size_t)ROWS_ * 3 * EMB_ * 2);
    float*  v_f     = (float*) carve((size_t)ROWS_ * EMB_ * 4);
    float*  xdk     = (float*) carve((size_t)ROWS_ * 4);
    float*  xdq     = (float*) carve((size_t)ROWS_ * 4);
    bf16_t* kp_b    = (bf16_t*)carve((size_t)ROWS_ * M_ * 2);
    bf16_t* qp_b    = (bf16_t*)carve((size_t)ROWS_ * M_ * 2);
    float*  ksum    = (float*) carve((size_t)B_ * M_ * 4);
    float*  dinv    = (float*) carve((size_t)ROWS_ * 4);
    bf16_t* vT_b    = (bf16_t*)carve((size_t)ROWS_ * EMB_ * 2);   // [b,n,t]
    bf16_t* kpT_b   = (bf16_t*)carve((size_t)ROWS_ * M_ * 2);     // [b,m,t]
    bf16_t* kptv_b  = (bf16_t*)carve((size_t)B_ * EMB_ * M_ * 2); // [b,n,m]
    bf16_t* yatt_b  = (bf16_t*)carve((size_t)ROWS_ * EMB_ * 2);
    float*  y_f     = (float*) carve((size_t)ROWS_ * EMB_ * 4);
    bf16_t* yn_b    = (bf16_t*)carve((size_t)ROWS_ * EMB_ * 2);
    bf16_t* h_b     = (bf16_t*)carve((size_t)ROWS_ * EMB_ * 2);

    auto cvt = [&](const float* src, bf16_t* dst, int n) {
        f2bf_kernel<<<(n + 255) / 256, 256, 0, stream>>>(src, dst, n);
    };
    cvt(w_kqv,  wkqv_b,  3 * EMB_ * DIM_);
    cvt(w_perf, wperf_b, M_ * EMB_);
    cvt(w_proj, wproj_b, EMB_ * EMB_);
    cvt(w_mlp1, wm1_b,   EMB_ * EMB_);
    cvt(w_mlp2, wm2_b,   EMB_ * EMB_);

    // LN1
    ln_kernel<<<ROWS_ / 8, 256, 0, stream>>>(x, g1, be1, xn_b, DIM_);

    // kqv = xn·w_kqv^T + b  (v side-stored fp32)
    gemm_kernel<MODE_KQV><<<dim3(3 * EMB_ / 128, ROWS_ / 128), 256, 0, stream>>>(
        xn_b, DIM_, wkqv_b, DIM_, 0, b_kqv, DIM_, kqv_b, 3 * EMB_, v_f, nullptr, nullptr);

    // xd_k, xd_q
    sumsq_kernel<<<ROWS_ / 8, 256, 0, stream>>>(kqv_b, xdk, xdq);

    // kp, qp
    gemm_kernel<MODE_PRM><<<dim3(M_ / 128, ROWS_ / 128), 256, 0, stream>>>(
        kqv_b, 3 * EMB_, wperf_b, EMB_, 0, nullptr, EMB_, kp_b, M_, nullptr, xdk, nullptr);
    gemm_kernel<MODE_PRM><<<dim3(M_ / 128, ROWS_ / 128), 256, 0, stream>>>(
        kqv_b + EMB_, 3 * EMB_, wperf_b, EMB_, 0, nullptr, EMB_, qp_b, M_, nullptr, xdq, nullptr);

    // ksum, Dinv
    ksum_kernel<<<B_ * M_ / 256, 256, 0, stream>>>(kp_b, ksum);
    dinv_kernel<<<ROWS_ / 8, 256, 0, stream>>>(qp_b, ksum, dinv);

    // transpose v (from kqv cols 1024..1535) and kp into K-contiguous layouts
    transpose_bf16_kernel<<<dim3(EMB_ / 32, T_ / 32, B_), 256, 0, stream>>>(
        kqv_b + 2 * EMB_, (long)T_ * 3 * EMB_, 3 * EMB_, vT_b, (long)EMB_ * T_, T_);
    transpose_bf16_kernel<<<dim3(M_ / 32, T_ / 32, B_), 256, 0, stream>>>(
        kp_b, (long)T_ * M_, M_, kpT_b, (long)M_ * T_, T_);

    // kptv[b] = vT[b]·kpT[b]^T  (normal K-contiguous GEMM per batch)
    for (int b = 0; b < B_; b++) {
        gemm_kernel<MODE_PLAIN><<<dim3(M_ / 128, EMB_ / 128), 256, 0, stream>>>(
            vT_b + (long)b * EMB_ * T_, T_, kpT_b + (long)b * M_ * T_, T_, 0,
            nullptr, T_, kptv_b + (long)b * EMB_ * M_, M_, nullptr, nullptr, nullptr);
    }

    // y_att = (qp·kptv^T) * Dinv   (per-batch weight via wBatchStride)
    gemm_kernel<MODE_ATT><<<dim3(EMB_ / 128, ROWS_ / 128), 256, 0, stream>>>(
        qp_b, M_, kptv_b, M_, (long)EMB_ * M_, nullptr, M_, yatt_b, EMB_, nullptr, dinv, nullptr);

    // y = v + y_att·w_proj^T + b_proj (fp32)
    gemm_kernel<MODE_PROJ><<<dim3(EMB_ / 128, ROWS_ / 128), 256, 0, stream>>>(
        yatt_b, EMB_, wproj_b, EMB_, 0, b_proj, EMB_, nullptr, EMB_, y_f, nullptr, v_f);

    // LN2
    ln_kernel<<<ROWS_ / 8, 256, 0, stream>>>(y_f, g2, be2, yn_b, EMB_);

    // h = gelu(yn·w_mlp1^T + b_mlp1)
    gemm_kernel<MODE_MLP1><<<dim3(EMB_ / 128, ROWS_ / 128), 256, 0, stream>>>(
        yn_b, EMB_, wm1_b, EMB_, 0, b_mlp1, EMB_, h_b, EMB_, nullptr, nullptr, nullptr);

    // out = y + h·w_mlp2^T + b_mlp2 (fp32)
    gemm_kernel<MODE_MLP2><<<dim3(EMB_ / 128, ROWS_ / 128), 256, 0, stream>>>(
        h_b, EMB_, wm2_b, EMB_, 0, b_mlp2, EMB_, nullptr, EMB_, out, nullptr, y_f);
}
